// MultiHeadAttn_42159398978166
// MI455X (gfx1250) — matrix operand order repaired and checked by reading, not compiled, not run on hardware
//
#include <hip/hip_runtime.h>
#include <hip/hip_bf16.h>
#include <math.h>

typedef __bf16 bf16;
typedef __bf16 v16bf __attribute__((ext_vector_type(16)));
typedef float  v8f   __attribute__((ext_vector_type(8)));
typedef int    v4i   __attribute__((ext_vector_type(4)));

#define DIM   256
#define NHEAD 8
#define SEQ   2048
#define BATCH 2
// M_total = BATCH*SEQ = 4096 ; E = DIM*NHEAD = 2048

#if defined(__HIP_DEVICE_COMPILE__) && __has_builtin(__builtin_amdgcn_global_load_async_to_lds_b128)
#define HAS_ASYNC_LDS 1
#else
#define HAS_ASYNC_LDS 0
#endif

static __device__ __forceinline__ void async_copy_b128(void* lds_dst, const void* gsrc) {
#if HAS_ASYNC_LDS
  __builtin_amdgcn_global_load_async_to_lds_b128(
      (__attribute__((address_space(1))) v4i*)(v4i*)(void*)gsrc,
      (__attribute__((address_space(3))) v4i*)(v4i*)lds_dst, 0, 0);
#else
  *(uint4*)lds_dst = *(const uint4*)gsrc;
#endif
}

static __device__ __forceinline__ void async_wait0() {
#if HAS_ASYNC_LDS
#if __has_builtin(__builtin_amdgcn_s_wait_asynccnt)
  __builtin_amdgcn_s_wait_asynccnt(0);
#else
  asm volatile("s_wait_asynccnt 0" ::: "memory");
#endif
#endif
}

static __device__ __forceinline__ v8f v8f_zero() {
  v8f z = {0.f,0.f,0.f,0.f,0.f,0.f,0.f,0.f};
  return z;
}

// ---- WMMA fragment loaders (contiguous k = 16*half + e order, wave32) ----

// A-matrix 16x32 bf16, LDS stored row-major [m][k] with stride ld.
// lane = m + 16*half ; elem e -> K = 16*half + e
static __device__ __forceinline__ v16bf frag_a_lds(const bf16* base, int ld, int k0) {
  int lane = threadIdx.x & 31;
  int m = lane & 15, h = lane >> 4;
  const bf16* p = base + m * ld + k0;
  v16bf a;
#pragma unroll
  for (int e = 0; e < 8; ++e) a[e] = p[16 * h + e];
#pragma unroll
  for (int e = 0; e < 8; ++e) a[8 + e] = p[16 * h + 8 + e];
  return a;
}

// B-matrix 32x16 bf16 where LDS holds one row per output column: [n][k], stride ld.
// lane = n + 16*half ; elem e -> K = 16*half + e , N = n
static __device__ __forceinline__ v16bf frag_b_nk(const bf16* base, int ld, int k0) {
  int lane = threadIdx.x & 31;
  int n = lane & 15, h = lane >> 4;
  const bf16* p = base + n * ld + k0 + 16 * h;
  v16bf b;
#pragma unroll
  for (int e = 0; e < 16; ++e) b[e] = p[e];
  return b;
}

// B-matrix 32x16 bf16 where LDS holds [k][n] (V tile), stride ld, column base n0.
static __device__ __forceinline__ v16bf frag_b_kn(const bf16* base, int ld, int n0) {
  int lane = threadIdx.x & 31;
  int n = lane & 15, h = lane >> 4;
  const bf16* p = base + (16 * h) * ld + n0 + n;
  v16bf b;
#pragma unroll
  for (int e = 0; e < 16; ++e) b[e] = p[e * ld];
  return b;
}

static __device__ __forceinline__ v8f wmma_bf16(v16bf a, v16bf b, v8f c) {
  return __builtin_amdgcn_wmma_f32_16x16x32_bf16(false, a, false, b, (short)0, c, false, false);
}

// ---------------------------------------------------------------------------
// Kernel 1: QKV projection.  C[4096x2048] = x[4096x256] * W[2048x256]^T
// 128x128 tile / block (8 waves, wave tile 32x64). Scatter bf16 to [b,h,s,d].
// ---------------------------------------------------------------------------
__global__ __launch_bounds__(256)
void qkv_gemm_kernel(const float* __restrict__ x,
                     const float* __restrict__ Wq, const float* __restrict__ Wk,
                     const float* __restrict__ Wv,
                     bf16* __restrict__ qb, bf16* __restrict__ kb, bf16* __restrict__ vb) {
  const int which = blockIdx.z;
  const float* W = (which == 0) ? Wq : (which == 1) ? Wk : Wv;
  bf16* dst      = (which == 0) ? qb : (which == 1) ? kb : vb;
  const float scale = (which == 2) ? 1.0f : 0.25f;   // 256^(-1/4)

  __shared__ bf16 As[128 * 32];
  __shared__ bf16 Bs[128 * 32];

  const int Mbase = blockIdx.x * 128;
  const int Nbase = blockIdx.y * 128;
  const int tid = threadIdx.x;
  const int wid = tid >> 5;
  const int wm = (wid & 3) * 32;   // wave M offset
  const int wn = (wid >> 2) * 64;  // wave N offset

  v8f acc[2][4];
#pragma unroll
  for (int i = 0; i < 2; ++i)
#pragma unroll
    for (int j = 0; j < 4; ++j) acc[i][j] = v8f_zero();

  for (int k0 = 0; k0 < 256; k0 += 32) {
    __syncthreads();
    // stage A tile: 128x32 f32 -> bf16 (1024 float4, 4 per thread)
#pragma unroll
    for (int i = 0; i < 4; ++i) {
      int f = tid + i * 256;
      int r = f >> 3, c4 = (f & 7) * 4;
      const float* gp = x + (size_t)(Mbase + r) * 256 + k0 + c4;
      const float4 v = *(const float4*)gp;
      if (k0 + 32 < 256) __builtin_prefetch(gp + 32, 0, 3);
      bf16* s = As + r * 32 + c4;
      s[0] = (bf16)v.x; s[1] = (bf16)v.y; s[2] = (bf16)v.z; s[3] = (bf16)v.w;
    }
    // stage B tile: rows Nbase..Nbase+127 of W over K chunk
#pragma unroll
    for (int i = 0; i < 4; ++i) {
      int f = tid + i * 256;
      int r = f >> 3, c4 = (f & 7) * 4;
      const float* gp = W + (size_t)(Nbase + r) * 256 + k0 + c4;
      const float4 v = *(const float4*)gp;
      if (k0 + 32 < 256) __builtin_prefetch(gp + 32, 0, 3);
      bf16* s = Bs + r * 32 + c4;
      s[0] = (bf16)v.x; s[1] = (bf16)v.y; s[2] = (bf16)v.z; s[3] = (bf16)v.w;
    }
    __syncthreads();
#pragma unroll
    for (int mt = 0; mt < 2; ++mt) {
      v16bf a = frag_a_lds(As + (wm + mt * 16) * 32, 32, 0);
#pragma unroll
      for (int nt = 0; nt < 4; ++nt) {
        v16bf b = frag_b_nk(Bs + (wn + nt * 16) * 32, 32, 0);
        acc[mt][nt] = wmma_bf16(a, b, acc[mt][nt]);
      }
    }
  }

  // C layout: VGPR r holds M = r + 8*half, N = lane%16. Scatter to [b,h,s,d].
  const int lane = tid & 31, n = lane & 15, h = lane >> 4;
#pragma unroll
  for (int mt = 0; mt < 2; ++mt)
#pragma unroll
    for (int nt = 0; nt < 4; ++nt)
#pragma unroll
      for (int r = 0; r < 8; ++r) {
        int gi = Mbase + wm + mt * 16 + r + 8 * h;   // b*2048 + s
        int gj = Nbase + wn + nt * 16 + n;           // head*256 + d
        int b_ = gi >> 11, s_ = gi & 2047;
        int h_ = gj >> 8,  d_ = gj & 255;
        dst[(((size_t)(b_ * NHEAD + h_)) * SEQ + s_) * DIM + d_] =
            (bf16)(acc[mt][nt][r] * scale);
      }
}

// ---------------------------------------------------------------------------
// Kernel 2: flash attention per (b,h). 128 threads = 4 waves, 16 q-rows each.
// 64 KB LDS: Q 64x256 (32K) + K 32x256 (16K) + V 32x256 (16K); P overlays K.
// Staging uses GLOBAL_LOAD_ASYNC_TO_LDS_B128 (ASYNCcnt) when available.
// ---------------------------------------------------------------------------
__global__ __launch_bounds__(128)
void attn_kernel(const bf16* __restrict__ qb, const bf16* __restrict__ kb,
                 const bf16* __restrict__ vb, bf16* __restrict__ ab) {
  __shared__ bf16 Qs[64 * 256];
  __shared__ bf16 Ks[32 * 256];
  __shared__ bf16 Vs[32 * 256];

  const int bh = blockIdx.y;          // b*8 + h
  const int q0 = blockIdx.x * 64;
  const size_t head_off = (size_t)bh * SEQ * DIM;
  const bf16* Q = qb + head_off;
  const bf16* K = kb + head_off;
  const bf16* V = vb + head_off;

  const int tid = threadIdx.x;
  const int wid = tid >> 5;
  const int lane = tid & 31;
  const int n = lane & 15, h = lane >> 4;

  // stage Q block once: 64x256 bf16 = 2048 x b128, 16 per thread (async)
  {
    const bf16* src = Q + (size_t)q0 * 256;
#pragma unroll
    for (int i = 0; i < 16; ++i) {
      int e = (tid + i * 128) * 8;     // bf16 element index
      async_copy_b128(Qs + e, src + e);
    }
  }

  v8f o[16];                       // O[16 x 256] accumulator (16 n-tiles)
#pragma unroll
  for (int nt = 0; nt < 16; ++nt) o[nt] = v8f_zero();
  float m_i[8], l_i[8];            // per lane: rows r + 8*half
#pragma unroll
  for (int r = 0; r < 8; ++r) { m_i[r] = -1e30f; l_i[r] = 0.f; }

  for (int kblk = 0; kblk < SEQ; kblk += 32) {
    __syncthreads();                 // previous iteration done with Ks/Vs
    // stage K,V blocks: 16x256 bf16 (first 16 of 32 rows) = 512 x b128 each, 4 per thread (async)
    {
      const bf16* sk = K + (size_t)kblk * 256;
      const bf16* sv = V + (size_t)kblk * 256;
#pragma unroll
      for (int i = 0; i < 4; ++i) {
        int e = (tid + i * 128) * 8;
        async_copy_b128(Ks + e, sk + e);
        async_copy_b128(Vs + e, sv + e);
      }
    }
    async_wait0();                   // this wave's async copies landed in LDS
    __syncthreads();                 // everyone's copies landed

    // S[16x32] = Q_tile @ K^T  (scale already baked into q,k)
    v8f s0 = v8f_zero(), s1 = v8f_zero();
#pragma unroll
    for (int kc = 0; kc < 256; kc += 32) {
      v16bf aq = frag_a_lds(Qs + (wid * 16) * 256, 256, kc);
      v16bf b0 = frag_b_nk(Ks, 256, kc);            // keys 0..15
      v16bf b1 = frag_b_nk(Ks + 16 * 256, 256, kc); // keys 16..31
      s0 = wmma_bf16(aq, b0, s0);
      s1 = wmma_bf16(aq, b1, s1);
    }

    // online softmax: row reductions across the 16 lanes of each half
    float alpha[8];
#pragma unroll
    for (int r = 0; r < 8; ++r) {
      float t = fmaxf(s0[r], s1[r]);
#pragma unroll
      for (int off = 1; off < 16; off <<= 1) t = fmaxf(t, __shfl_xor(t, off, 32));
      float mn = fmaxf(m_i[r], t);
      alpha[r] = __expf(m_i[r] - mn);
      m_i[r] = mn;
      s0[r] = __expf(s0[r] - mn);
      s1[r] = __expf(s1[r] - mn);
      float rs = s0[r] + s1[r];
#pragma unroll
      for (int off = 1; off < 16; off <<= 1) rs += __shfl_xor(rs, off, 32);
      l_i[r] = l_i[r] * alpha[r] + rs;
    }
    // rescale O
#pragma unroll
    for (int nt = 0; nt < 16; ++nt)
#pragma unroll
      for (int r = 0; r < 8; ++r) o[nt][r] *= alpha[r];

    // transpose P through per-wave LDS scratch (overlaid on K region)
    __syncthreads();                       // all waves done reading Ks
    bf16* Ps = Ks + wid * (16 * 32);
#pragma unroll
    for (int r = 0; r < 8; ++r) {
      Ps[(r + 8 * h) * 32 + n]      = (bf16)s0[r];
      Ps[(r + 8 * h) * 32 + 16 + n] = (bf16)s1[r];
    }
    asm volatile("s_wait_dscnt 0" ::: "memory");
    v16bf pa = frag_a_lds(Ps, 32, 0);

    // O += P @ V  (V tile is [k][n] in LDS)
#pragma unroll
    for (int nt = 0; nt < 16; ++nt) {
      v16bf bv = frag_b_kn(Vs, 256, nt * 16);
      o[nt] = wmma_bf16(pa, bv, o[nt]);
    }
  }

  // normalize and store bf16 to [b][s][h*DIM + d]
  const int b_ = bh >> 3, h_ = bh & 7;
#pragma unroll
  for (int r = 0; r < 8; ++r) {
    float inv = 1.0f / l_i[r];
    int srow = q0 + wid * 16 + r + 8 * h;
    bf16* drow = ab + ((size_t)b_ * SEQ + srow) * (NHEAD * DIM) + h_ * DIM;
#pragma unroll
    for (int nt = 0; nt < 16; ++nt)
      drow[nt * 16 + n] = (bf16)(o[nt][r] * inv);
  }
}

// ---------------------------------------------------------------------------
// Kernel 3: output projection. out[4096x256] = A[4096x2048] * Wo[256x2048]^T + bo
// A tile staged with async global->LDS copies (pure bf16, no conversion).
// ---------------------------------------------------------------------------
__global__ __launch_bounds__(256)
void out_gemm_kernel(const bf16* __restrict__ A, const float* __restrict__ Wo,
                     const float* __restrict__ bo, float* __restrict__ out) {
  __shared__ bf16 As[128 * 32];
  __shared__ bf16 Bs[128 * 32];
  const int Mbase = blockIdx.x * 128;
  const int Nbase = blockIdx.y * 128;
  const int tid = threadIdx.x;
  const int wid = tid >> 5;
  const int wm = (wid & 3) * 32;
  const int wn = (wid >> 2) * 64;

  v8f acc[2][4];
#pragma unroll
  for (int i = 0; i < 2; ++i)
#pragma unroll
    for (int j = 0; j < 4; ++j) acc[i][j] = v8f_zero();

  for (int k0 = 0; k0 < 2048; k0 += 32) {
    __syncthreads();
    // stage A: 128x32 bf16 = 512 x b128, 2 per thread (async, no conversion)
#pragma unroll
    for (int i = 0; i < 2; ++i) {
      int f = tid + i * 256;
      int r = f >> 2, c8 = (f & 3) * 8;
      const bf16* gp = A + (size_t)(Mbase + r) * 2048 + k0 + c8;
      async_copy_b128(As + r * 32 + c8, gp);
      if (k0 + 32 < 2048) __builtin_prefetch(gp + 32, 0, 3);
    }
    // stage B: Wo rows f32 -> bf16, 128x32 = 1024 float4, 4 per thread
#pragma unroll
    for (int i = 0; i < 4; ++i) {
      int f = tid + i * 256;
      int r = f >> 3, c4 = (f & 7) * 4;
      const float* gp = Wo + (size_t)(Nbase + r) * 2048 + k0 + c4;
      const float4 v = *(const float4*)gp;
      if (k0 + 32 < 2048) __builtin_prefetch(gp + 32, 0, 3);
      bf16* s = Bs + r * 32 + c4;
      s[0] = (bf16)v.x; s[1] = (bf16)v.y; s[2] = (bf16)v.z; s[3] = (bf16)v.w;
    }
    async_wait0();
    __syncthreads();
#pragma unroll
    for (int mt = 0; mt < 2; ++mt) {
      v16bf a = frag_a_lds(As + (wm + mt * 16) * 32, 32, 0);
#pragma unroll
      for (int nt = 0; nt < 4; ++nt) {
        v16bf b = frag_b_nk(Bs + (wn + nt * 16) * 32, 32, 0);
        acc[mt][nt] = wmma_bf16(a, b, acc[mt][nt]);
      }
    }
  }

  const int lane = tid & 31, n = lane & 15, h = lane >> 4;
#pragma unroll
  for (int mt = 0; mt < 2; ++mt)
#pragma unroll
    for (int nt = 0; nt < 4; ++nt)
#pragma unroll
      for (int r = 0; r < 8; ++r) {
        int gi = Mbase + wm + mt * 16 + r + 8 * h;
        int gj = Nbase + wn + nt * 16 + n;
        out[(size_t)gi * 256 + gj] = acc[mt][nt][r] + bo[gj];
      }
}

// ---------------------------------------------------------------------------
extern "C" void kernel_launch(void* const* d_in, const int* in_sizes, int n_in,
                              void* d_out, int out_size, void* d_ws, size_t ws_size,
                              hipStream_t stream) {
  const float* x  = (const float*)d_in[0];
  const float* Wq = (const float*)d_in[1];
  const float* Wk = (const float*)d_in[2];
  const float* Wv = (const float*)d_in[3];
  const float* Wo = (const float*)d_in[4];
  const float* bo = (const float*)d_in[5];
  float* out = (float*)d_out;

  char* ws = (char*)d_ws;
  const size_t qkv_bytes = (size_t)BATCH * NHEAD * SEQ * DIM * sizeof(bf16); // 16 MB
  bf16* qb = (bf16*)(ws);
  bf16* kb = (bf16*)(ws + qkv_bytes);
  bf16* vb = (bf16*)(ws + 2 * qkv_bytes);
  bf16* ab = (bf16*)(ws + 3 * qkv_bytes);

  dim3 g1(4096 / 128, 2048 / 128, 3);
  qkv_gemm_kernel<<<g1, 256, 0, stream>>>(x, Wq, Wk, Wv, qb, kb, vb);

  dim3 g2(SEQ / 64, BATCH * NHEAD);
  attn_kernel<<<g2, 128, 0, stream>>>(qb, kb, vb, ab);

  dim3 g3(4096 / 128, 256 / 128);
  out_gemm_kernel<<<g3, 256, 0, stream>>>(ab, Wo, bo, out);
}
